// SUGPool_27109833572930
// MI455X (gfx1250) — compile-verified, matrix operand order checked by reading, not run on hardware
//
#include <hip/hip_runtime.h>
#include <hip/hip_bf16.h>
#include <math.h>

typedef __attribute__((ext_vector_type(16))) _Float16 v16h;
typedef __attribute__((ext_vector_type(8)))  float    v8f;

#define CCH 128          // channels
#define COMBINE 0.5f

static __host__ __device__ inline int cdiv_i(int a, int b) { return (a + b - 1) / b; }

// ---------------------------------------------------------------------------
// Kernel: fill float region
// ---------------------------------------------------------------------------
__global__ void fill_f32_kernel(float* __restrict__ p, int n, float v) {
    int i = blockIdx.x * blockDim.x + threadIdx.x;
    if (i < n) p[i] = v;
}

__global__ void fill_i32_kernel(int* __restrict__ p, int n, int v) {
    int i = blockIdx.x * blockDim.x + threadIdx.x;
    if (i < n) p[i] = v;
}

// ---------------------------------------------------------------------------
// Kernel 1: kv = relu(x @ W^T + b) -- GEMV via V_WMMA_F32_16X16X32_F16.
// One wave computes 16 rows. A = 16x32 f16 tile of x (lane=row, half-wave
// selects K sub-block per ISA 7.12.2). B = 32x16 with W only in column N=0
// (lanes 0 and 16). Column 0 of D holds the 16 dot products.
// ---------------------------------------------------------------------------
__global__ void kv_wmma_kernel(const float* __restrict__ x,
                               const float* __restrict__ W,
                               const float* __restrict__ bias_p,
                               float* __restrict__ kv,
                               int nWaves) {
    int tid  = blockIdx.x * blockDim.x + threadIdx.x;
    int wid  = tid >> 5;                 // uniform across wave
    if (wid >= nWaves) return;           // wave-uniform: EXEC stays all-1s
    int lane = threadIdx.x & 31;
    int r    = (wid << 4) + (lane & 15); // row this lane slot feeds into A
    int h    = lane >> 4;                // which K half-block
    bool n0  = (lane & 15) == 0;         // lanes holding column N=0 of B/D

    const float* xr = x + (size_t)r * CCH;

    v8f acc = {};
#pragma unroll
    for (int kc = 0; kc < CCH; kc += 32) {
        v16h a, bm;
        int base0 = kc + 8 * h;          // halves 0..7  -> K = base0..base0+7
        int base1 = kc + 16 + 8 * h;     // halves 8..15 -> K = base1..base1+7
#pragma unroll
        for (int j = 0; j < 8; ++j) {
            a[j]      = (_Float16)xr[base0 + j];
            a[8 + j]  = (_Float16)xr[base1 + j];
            float w0  = W[base0 + j];
            float w1  = W[base1 + j];
            bm[j]     = n0 ? (_Float16)w0 : (_Float16)0.0f;
            bm[8 + j] = n0 ? (_Float16)w1 : (_Float16)0.0f;
        }
        acc = __builtin_amdgcn_wmma_f32_16x16x32_f16(false, a, false, bm,
                                                     (short)0, acc, false, false);
    }
    // D layout (32-bit C/D 16x16): lane 0 -> M=0..7 (VGPR 0..7), lane 16 -> M=8..15
    if (n0) {
        float bias = bias_p[0];
        int r0 = (wid << 4) + 8 * h;
#pragma unroll
        for (int v = 0; v < 8; ++v) {
            float s = acc[v] + bias;
            kv[r0 + v] = s > 0.0f ? s : 0.0f;
        }
    }
}

// ---------------------------------------------------------------------------
// Kernel 2: edge pass 1: deg[row] += w ; s[row] += w * kv[col]
// ---------------------------------------------------------------------------
__global__ void edge_pass1_kernel(const int* __restrict__ row,
                                  const int* __restrict__ col,
                                  const float* __restrict__ ea,
                                  const float* __restrict__ kv,
                                  float* __restrict__ s,
                                  float* __restrict__ deg,
                                  int E) {
    int e = blockIdx.x * blockDim.x + threadIdx.x;
    if (e >= E) return;
    int r = row[e], c = col[e];
    float w = ea[e];
    atomicAdd(&deg[r], w);
    atomicAdd(&s[r], w * kv[c]);
}

// ---------------------------------------------------------------------------
// Kernel 3: node pass 1: score + deg_inv + init sort (key,idx) incl. padding
// ---------------------------------------------------------------------------
__global__ void node_pass1_kernel(const float* __restrict__ s,
                                  const float* __restrict__ kv,
                                  const float* __restrict__ deg,
                                  float* __restrict__ deg_inv,
                                  float* __restrict__ skey,
                                  int* __restrict__ sidx,
                                  int N, int NPAD) {
    int i = blockIdx.x * blockDim.x + threadIdx.x;
    if (i >= NPAD) return;
    if (i < N) {
        skey[i] = s[i] + kv[i];
        sidx[i] = i;
        float d = deg[i];
        deg_inv[i] = d > 0.0f ? 1.0f / d : 0.0f;
    } else {
        skey[i] = -INFINITY;
        sidx[i] = 0x7FFFFFFF;
    }
}

// ---------------------------------------------------------------------------
// Bitonic sort pass: order = descending key, ties -> lower index first
// ---------------------------------------------------------------------------
__device__ inline bool sort_before(float ka, int ia, float kb, int ib) {
    return (ka > kb) || (ka == kb && ia < ib);
}

__global__ void bitonic_pass_kernel(float* __restrict__ skey,
                                    int* __restrict__ sidx,
                                    unsigned k, unsigned j, int NPAD) {
    unsigned i = blockIdx.x * blockDim.x + threadIdx.x;
    if ((int)i >= NPAD) return;
    unsigned ixj = i ^ j;
    if (ixj <= i) return;
    float ka = skey[i],  kb = skey[ixj];
    int   ia = sidx[i],  ib = sidx[ixj];
    bool up = ((i & k) == 0);
    bool swap = up ? sort_before(kb, ib, ka, ia)     // want a before b
                   : sort_before(ka, ia, kb, ib);    // reversed run
    if (swap) {
        skey[i] = kb; skey[ixj] = ka;
        sidx[i] = ib; sidx[ixj] = ia;
    }
}

// ---------------------------------------------------------------------------
// Kernel 4: edge pass 2 (one wave per edge, 4 channels per lane):
//   wn = w * deg_inv[row]; w2 = wn * kv[col]
//   deg2[row] += wn ; xn[row][:] += w2 * x[col][:]
// ---------------------------------------------------------------------------
__global__ void edge_pass2_kernel(const int* __restrict__ row,
                                  const int* __restrict__ col,
                                  const float* __restrict__ ea,
                                  const float* __restrict__ kv,
                                  const float* __restrict__ deg_inv,
                                  const float* __restrict__ x,
                                  float* __restrict__ deg2,
                                  float* __restrict__ xn,
                                  int E) {
    int t  = blockIdx.x * blockDim.x + threadIdx.x;
    int we = t >> 5;                     // one wave per edge (uniform in wave)
    if (we >= E) return;
    int lane = threadIdx.x & 31;

    int r = row[we], c = col[we];
    float wn = ea[we] * deg_inv[r];
    float w2 = wn * kv[c];
    if (lane == 0) atomicAdd(&deg2[r], wn);

    // prefetch a future edge's source row to hide indirect-gather latency
    int ahead = we + 64;
    if (ahead < E) {
        const float* pf = x + (size_t)col[ahead] * CCH + lane * 4;
        __builtin_prefetch(pf, 0, 1);    // -> global_prefetch_b8
    }

    const float4* xc = (const float4*)(x + (size_t)c * CCH);
    float4 v = xc[lane];
    float* dst = xn + (size_t)r * CCH + lane * 4;
    atomicAdd(dst + 0, w2 * v.x);
    atomicAdd(dst + 1, w2 * v.y);
    atomicAdd(dst + 2, w2 * v.z);
    atomicAdd(dst + 3, w2 * v.w);
}

// ---------------------------------------------------------------------------
// Kernel 5: node combine (in place into xn):
//   xn[i][c] = xn[i][c]*deg2_inv[i]*(1-COMBINE) + x[i][c]*kv[i]*COMBINE
// ---------------------------------------------------------------------------
__global__ void node_combine_kernel(const float* __restrict__ x,
                                    const float* __restrict__ kv,
                                    const float* __restrict__ deg2,
                                    float* __restrict__ xn,
                                    int total) {
    int t = blockIdx.x * blockDim.x + threadIdx.x;
    if (t >= total) return;
    int i = t >> 7;                      // node (CCH == 128)
    float d2 = deg2[i];
    float inv = d2 > 0.0f ? 1.0f / d2 : 0.0f;
    xn[t] = xn[t] * inv * (1.0f - COMBINE) + x[t] * kv[i] * COMBINE;
}

// ---------------------------------------------------------------------------
// Kernel 6: top-K scatter: perm, batch, idx_map[perm[j]] = j
// ---------------------------------------------------------------------------
__global__ void topk_scatter_kernel(const int* __restrict__ sidx,
                                    int* __restrict__ perm_out,
                                    int* __restrict__ batch_out,
                                    int* __restrict__ idx_map,
                                    int K) {
    int j = blockIdx.x * blockDim.x + threadIdx.x;
    if (j >= K) return;
    int p = sidx[j];
    perm_out[j]  = p;
    batch_out[j] = 0;
    idx_map[p]   = j;
}

// ---------------------------------------------------------------------------
// Kernel 7: gather x_out[j][:] = xn[perm[j]][:]
// ---------------------------------------------------------------------------
__global__ void gather_xout_kernel(const float* __restrict__ xn,
                                   const int* __restrict__ sidx,
                                   float* __restrict__ xout,
                                   int total) {
    int t = blockIdx.x * blockDim.x + threadIdx.x;
    if (t >= total) return;
    int j  = t >> 7;
    int ch = t & (CCH - 1);
    xout[t] = xn[(size_t)sidx[j] * CCH + ch];
}

// ---------------------------------------------------------------------------
// Kernel 8: edge remap through idx_map
// ---------------------------------------------------------------------------
__global__ void edge_remap_kernel(const int* __restrict__ row,
                                  const int* __restrict__ col,
                                  const float* __restrict__ ea,
                                  const int* __restrict__ idx_map,
                                  int* __restrict__ out_ei,
                                  float* __restrict__ out_ea,
                                  int E) {
    int e = blockIdx.x * blockDim.x + threadIdx.x;
    if (e >= E) return;
    int nr = idx_map[row[e]];
    int nc = idx_map[col[e]];
    bool m = (nr >= 0) && (nc >= 0);
    out_ei[e]     = m ? nr : -1;
    out_ei[E + e] = m ? nc : -1;
    out_ea[e]     = m ? ea[e] : 0.0f;
}

// ---------------------------------------------------------------------------
// Host launcher
// ---------------------------------------------------------------------------
extern "C" void kernel_launch(void* const* d_in, const int* in_sizes, int n_in,
                              void* d_out, int out_size, void* d_ws, size_t ws_size,
                              hipStream_t stream) {
    const float* x   = (const float*)d_in[0];
    const int*   ei  = (const int*)  d_in[1];
    const float* ea  = (const float*)d_in[2];
    const float* W   = (const float*)d_in[3];
    const float* bia = (const float*)d_in[4];

    const int N = in_sizes[0] / CCH;
    const int E = in_sizes[2];
    const int K = (int)((4LL * N + 4) / 5);      // ceil(0.8*N)
    int NPAD = 1; while (NPAD < N) NPAD <<= 1;

    const int* row = ei;
    const int* col = ei + E;

    // ---- workspace layout (fp32 units) ----
    float* ws    = (float*)d_ws;
    float* kv    = ws;                 ws += N;
    float* s     = ws;                 ws += N;    // |
    float* deg   = ws;                 ws += N;    // | contiguous 3N, zeroed
    float* deg2  = ws;                 ws += N;    // |
    float* dinv  = ws;                 ws += N;
    float* skey  = ws;                 ws += NPAD;
    int*   sidx  = (int*)ws;           ws += NPAD;
    int*   imap  = (int*)ws;           ws += N;
    float* xn    = ws;                 ws += (size_t)N * CCH;

    // ---- output layout (flat, return order) ----
    float* xout      = (float*)d_out;                        // K*CCH f32
    int*   out_ei    = (int*)  (xout + (size_t)K * CCH);     // 2*E i32
    float* out_ea    = (float*)(out_ei + 2 * (size_t)E);     // E f32
    int*   out_batch = (int*)  (out_ea + E);                 // K i32
    int*   out_perm  = out_batch + K;                        // K i32

    const int B = 256;

    // 0) zero accumulators (graph replay: must re-init every call)
    fill_f32_kernel<<<cdiv_i(3 * N, B), B, 0, stream>>>(s, 3 * N, 0.0f);
    fill_f32_kernel<<<cdiv_i(N * CCH, B), B, 0, stream>>>(xn, N * CCH, 0.0f);
    fill_i32_kernel<<<cdiv_i(N, B), B, 0, stream>>>(imap, N, -1);

    // 1) kv via WMMA (one wave per 16 rows; N is a multiple of 16 here)
    int nWaves = cdiv_i(N, 16);
    kv_wmma_kernel<<<cdiv_i(nWaves * 32, B), B, 0, stream>>>(x, W, bia, kv, nWaves);

    // 2) edge pass 1: deg, s
    edge_pass1_kernel<<<cdiv_i(E, B), B, 0, stream>>>(row, col, ea, kv, s, deg, E);

    // 3) node pass 1: score / deg_inv / sort init (with padding)
    node_pass1_kernel<<<cdiv_i(NPAD, B), B, 0, stream>>>(s, kv, deg, dinv,
                                                         skey, sidx, N, NPAD);

    // 4) bitonic sort (descending score, lower index wins ties)
    for (unsigned k = 2; k <= (unsigned)NPAD; k <<= 1)
        for (unsigned j = k >> 1; j > 0; j >>= 1)
            bitonic_pass_kernel<<<cdiv_i(NPAD, B), B, 0, stream>>>(skey, sidx, k, j, NPAD);

    // 5) edge pass 2: neighbor aggregation (wave per edge)
    edge_pass2_kernel<<<cdiv_i(E * 32, B), B, 0, stream>>>(row, col, ea, kv, dinv,
                                                           x, deg2, xn, E);

    // 6) node combine (in place)
    node_combine_kernel<<<cdiv_i(N * CCH, B), B, 0, stream>>>(x, kv, deg2, xn, N * CCH);

    // 7) perm / batch / idx_map
    topk_scatter_kernel<<<cdiv_i(K, B), B, 0, stream>>>(sidx, out_perm, out_batch, imap, K);

    // 8) gather pooled features
    gather_xout_kernel<<<cdiv_i(K * CCH, B), B, 0, stream>>>(xn, sidx, xout, K * CCH);

    // 9) edge remap
    edge_remap_kernel<<<cdiv_i(E, B), B, 0, stream>>>(row, col, ea, imap,
                                                      out_ei, out_ea, E);
}
